// GraphConv_75110388073051
// MI455X (gfx1250) — matrix operand order repaired and checked by reading, not compiled, not run on hardware
//
#include <hip/hip_runtime.h>
#include <hip/hip_bf16.h>

// ---------------------------------------------------------------------------
// Problem: B=8, N=128, D=64, E=3, H=256, L=4, DEC_OUT=64
// Strategy:
//   h1[b,e,i,j,:] = relu(P[b,e,i,:] + Q[b,e,j,:])       (be1 folded into P)
//   with P = ns @ We1[:, :64, :] + be1,  Q = ns @ We1[:, 64:, :]
//   h2 = relu(h1 @ We2[e] + be2[e]); aggr[b,j] = sum_{e,i} mask(e,i,j)*h2
//   out = relu(relu([ns, aggr] @ Wd1 + bd1) @ Wd2 + bd2)
// Hot loop (51.5 GFLOP) runs on v_wmma_f32_16x16x32_bf16.
// ---------------------------------------------------------------------------

typedef __attribute__((ext_vector_type(16))) __bf16 v16bf;
typedef __attribute__((ext_vector_type(8)))  __bf16 v8bf;
typedef __attribute__((ext_vector_type(8)))  float  v8f;

#define Bq 8
#define Nq 128
#define Dq 64
#define Eq 3
#define Hq 256

// ---------------------------------------------------------------------------
// Kernel A: P[b,e,i,h] = be1[e,h] + sum_d ns[b,i,d]*We1[e,d,h]
//           Q[b,e,i,h] =            sum_d ns[b,i,d]*We1[e,64+d,h]
// grid (N, B, E), block 256 (one thread per h)
// ---------------------------------------------------------------------------
__global__ __launch_bounds__(256) void pq_kernel(const float* __restrict__ ns,
                                                 const float* __restrict__ We1,
                                                 const float* __restrict__ be1,
                                                 float* __restrict__ P,
                                                 float* __restrict__ Q) {
    const int i = blockIdx.x, b = blockIdx.y, e = blockIdx.z;
    const int h = threadIdx.x;
    __shared__ float sns[Dq];
    if (h < Dq) sns[h] = ns[(b * Nq + i) * Dq + h];
    __syncthreads();
    float s1 = be1[e * Hq + h];
    float s2 = 0.f;
    const float* w = We1 + (size_t)e * 2 * Dq * Hq + h;
    #pragma unroll 4
    for (int d = 0; d < Dq; ++d) {
        s1 = fmaf(sns[d], w[(size_t)d * Hq], s1);
        s2 = fmaf(sns[d], w[(size_t)(Dq + d) * Hq], s2);
    }
    const size_t o = ((size_t)(b * Eq + e) * Nq + i) * Hq + h;
    P[o] = s1;
    Q[o] = s2;
}

// ---------------------------------------------------------------------------
// Kernel W: shuffle We2 (E,256,256) f32 -> bf16 WMMA B-fragments.
// Fragment layout per (e, wave w, kstep kk): [lane(32)][t(16)] with
//   K = kk*32 + t + 16*(lane>=16), col = w*16 + (lane&15)
// so each lane's 16 bf16 (32B) are contiguous.
// ---------------------------------------------------------------------------
__global__ __launch_bounds__(256) void bfrag_kernel(const float* __restrict__ We2,
                                                    unsigned short* __restrict__ Bfrag) {
    const int gid = blockIdx.x * 256 + threadIdx.x;   // < 3*65536
    const int t    = gid & 15;
    const int lane = (gid >> 4) & 31;
    const int kk   = (gid >> 9) & 7;
    const int w    = (gid >> 12) & 15;
    const int e    = gid >> 16;
    const int K    = kk * 32 + t + 16 * (lane >> 4);
    const int col  = w * 16 + (lane & 15);
    __bf16 v = (__bf16)We2[((size_t)e * Hq + K) * Hq + col];
    Bfrag[gid] = __builtin_bit_cast(unsigned short, v);
}

// ---------------------------------------------------------------------------
// Kernel B: fused edge-MLP layer-2 (WMMA) + masked aggregation + decoder.
// One workgroup per (b,j); 512 threads = 16 waves; wave w owns columns
// [16w,16w+16). For each (e, i-block of 16): build h1 tile (16x256 bf16) in
// LDS, run 8x v_wmma_f32_16x16x32_bf16, bias+relu+edge-mask, reduce rows.
// ---------------------------------------------------------------------------
__global__ __launch_bounds__(512) void fused_kernel(const float* __restrict__ P,
                                                    const float* __restrict__ Q,
                                                    const unsigned short* __restrict__ BfragU,
                                                    const float* __restrict__ edges,
                                                    const float* __restrict__ ns,
                                                    const float* __restrict__ be2,
                                                    const float* __restrict__ Wd1,
                                                    const float* __restrict__ bd1,
                                                    const float* __restrict__ Wd2,
                                                    const float* __restrict__ bd2,
                                                    float* __restrict__ out) {
    const int j = blockIdx.x, b = blockIdx.y;
    const int tid  = threadIdx.x;
    const int lane = tid & 31;
    const int wid  = tid >> 5;          // 0..15
    const int n0   = wid * 16;
    const int col  = lane & 15;
    const int half8 = (lane >> 4) * 8;  // 0 or 8: row offset of this half-wave in C

    __shared__ float  qb[Eq * Hq];       // Q[b,e,j,:]
    __shared__ float  smask[Eq * Nq];    // edges[b,i,j,e+1]
    __shared__ __bf16 h1tile[16 * Hq];   // current 16x256 h1 tile (bf16)
    __shared__ float  decin[Dq + Hq];    // [ns(64) | aggr(256)]
    __shared__ float  dec1[Hq];

    // stage per-(b,j) constants
    for (int idx = tid; idx < Eq * Hq; idx += 512) {
        int e = idx >> 8, k = idx & 255;
        qb[idx] = Q[((size_t)(b * Eq + e) * Nq + j) * Hq + k];
    }
    if (tid < Eq * Nq) {
        int e = tid / Nq, i = tid % Nq;
        smask[tid] = edges[((size_t)(b * Nq + i) * Nq + j) * 4 + (e + 1)];
    }
    if (tid < Dq) decin[tid] = ns[(size_t)(b * Nq + j) * Dq + tid];

    const __bf16* Bfrag = reinterpret_cast<const __bf16*>(BfragU);
    float accv = 0.f;

    for (int e = 0; e < Eq; ++e) {
        const float be2v = be2[e * Hq + n0 + col];
        const float* qbe = qb + e * Hq;
        const float* me  = smask + e * Nq;
        for (int ib = 0; ib < 8; ++ib) {
            const int i0 = ib * 16;
            __syncthreads();
            // cooperative build of h1 tile: relu(P[i0+row,:] + Q[j,:]) -> bf16
            const float* Pt = P + ((size_t)(b * Eq + e) * Nq + i0) * Hq;
            #pragma unroll
            for (int t = tid; t < 16 * Hq; t += 512) {
                float v = Pt[t] + qbe[t & 255];
                h1tile[t] = (__bf16)fmaxf(v, 0.f);
            }
            __syncthreads();

            // WMMA chain: C(16x16) += A(16x32) * B(32x16) over K=256
            v8f c = {0.f, 0.f, 0.f, 0.f, 0.f, 0.f, 0.f, 0.f};
            const __bf16* hrow = h1tile + col * Hq;
            const v16bf* bptr =
                reinterpret_cast<const v16bf*>(Bfrag + (((size_t)(e * 16 + wid) * 8) << 9));
            #pragma unroll
            for (int kk = 0; kk < 8; ++kk) {
                const int off = kk * 32 + 2 * half8;
                v8bf lo = *reinterpret_cast<const v8bf*>(hrow + off);
                v8bf hi = *reinterpret_cast<const v8bf*>(hrow + off + 8);
                v16bf a = __builtin_shufflevector(lo, hi, 0, 1, 2, 3, 4, 5, 6, 7,
                                                  8, 9, 10, 11, 12, 13, 14, 15);
                v16bf bf = bptr[(size_t)kk * 32 + lane];
                c = __builtin_amdgcn_wmma_f32_16x16x32_bf16(
                        false, a, false, bf, (short)0, c, false, false);
            }
            // bias + relu + edge-type mask, reduce the 8 rows of this half
            #pragma unroll
            for (int r = 0; r < 8; ++r) {
                float h2 = fmaxf(c[r] + be2v, 0.f);
                accv = fmaf(h2, me[i0 + half8 + r], accv);
            }
        }
    }

    // fold the two row-halves (lane <-> lane^16) and publish aggr columns
    accv += __shfl_xor(accv, 16);
    if (lane < 16) decin[Dq + n0 + lane] = accv;
    __syncthreads();

    // decoder layer 1: 320 -> 256
    if (tid < Hq) {
        float s = bd1[tid];
        #pragma unroll 4
        for (int k = 0; k < Dq + Hq; ++k)
            s = fmaf(decin[k], Wd1[(size_t)k * Hq + tid], s);
        dec1[tid] = fmaxf(s, 0.f);
    }
    __syncthreads();

    // decoder layer 2: 256 -> 64
    if (tid < 64) {
        float s = bd2[tid];
        #pragma unroll 4
        for (int k = 0; k < Hq; ++k)
            s = fmaf(dec1[k], Wd2[(size_t)k * 64 + tid], s);
        out[((size_t)b * Nq + j) * 64 + tid] = fmaxf(s, 0.f);
    }
}

// ---------------------------------------------------------------------------
extern "C" void kernel_launch(void* const* d_in, const int* in_sizes, int n_in,
                              void* d_out, int out_size, void* d_ws, size_t ws_size,
                              hipStream_t stream) {
    (void)in_sizes; (void)n_in; (void)out_size; (void)ws_size;
    const float* ns   = (const float*)d_in[0];
    const float* edges = (const float*)d_in[1];
    const float* We1  = (const float*)d_in[2];
    const float* be1  = (const float*)d_in[3];
    const float* We2  = (const float*)d_in[4];
    const float* be2  = (const float*)d_in[5];
    const float* Wd1  = (const float*)d_in[6];
    const float* bd1  = (const float*)d_in[7];
    const float* Wd2  = (const float*)d_in[8];
    const float* bd2  = (const float*)d_in[9];
    float* out = (float*)d_out;

    char* ws = (char*)d_ws;
    const size_t PQ_BYTES = (size_t)Bq * Eq * Nq * Hq * sizeof(float); // 3 MiB
    float* P = (float*)ws;
    float* Q = (float*)(ws + PQ_BYTES);
    unsigned short* Bfrag = (unsigned short*)(ws + 2 * PQ_BYTES);      // 384 KiB

    pq_kernel<<<dim3(Nq, Bq, Eq), 256, 0, stream>>>(ns, We1, be1, P, Q);
    bfrag_kernel<<<(Eq * 65536) / 256, 256, 0, stream>>>(We2, Bfrag);
    fused_kernel<<<dim3(Nq, Bq), 512, 0, stream>>>(P, Q, Bfrag, edges, ns, be2,
                                                   Wd1, bd1, Wd2, bd2, out);
}
